// MultiPersLandscapeValLayer_73675868996036
// MI455X (gfx1250) — matrix operand order repaired and checked by reading, not compiled, not run on hardware
//
#include <hip/hip_runtime.h>
#include <hip/hip_bf16.h>
#include <math.h>

#define N_V 12288
#define DIM 32
#define N_E 196608
#define EPS_F 1e-12f
#define FLT_BIG 3.0e38f
#define LDS_STRIDE 20   // 80B rows: 16B-aligned row starts, conflict-free stores
#define NTILE 192       // (N_V/16)/4 column tiles per wave

typedef _Float16 v16h __attribute__((ext_vector_type(16)));
typedef _Float16 v8h  __attribute__((ext_vector_type(8)));
typedef float    v8f  __attribute__((ext_vector_type(8)));
typedef float    v4f  __attribute__((ext_vector_type(4)));

// ---------------- helpers ----------------

// Pop the head of whichever sorted 5-list (a*, b*) is smaller into dst.
#define PICK(dst)                                                          \
  do {                                                                     \
    if (a0 <= b0) { dst = a0; a0 = a1; a1 = a2; a2 = a3; a3 = a4; a4 = FLT_BIG; } \
    else          { dst = b0; b0 = b1; b1 = b2; b2 = b3; b3 = b4; b4 = FLT_BIG; } \
  } while (0)

// Merge two ascending 5-lists, keep the 5 smallest (ascending).
__device__ __forceinline__ void merge5(float a0, float a1, float a2, float a3, float a4,
                                       float b0, float b1, float b2, float b3, float b4,
                                       float& r0, float& r1, float& r2, float& r3, float& r4) {
  PICK(r0); PICK(r1); PICK(r2); PICK(r3); PICK(r4);
}

// Branchless sorted insert of d into ascending t0..t4 (keep 5 smallest).
__device__ __forceinline__ void ins5(float d, float& t0, float& t1, float& t2,
                                     float& t3, float& t4) {
  float y0 = fmaxf(t0, d);  t0 = fminf(t0, d);
  float y1 = fmaxf(t1, y0); t1 = fminf(t1, y0);
  float y2 = fmaxf(t2, y1); t2 = fminf(t2, y1);
  float y3 = fmaxf(t3, y2); t3 = fminf(t3, y2);
  t4 = fminf(t4, y3);
}

// Scan 8 candidates: min-tree reject test, rare branch into insertion network.
__device__ __forceinline__ void scan8(float c0, float c1, float c2, float c3,
                                      float c4, float c5, float c6, float c7,
                                      float& t0, float& t1, float& t2,
                                      float& t3, float& t4) {
  float mn = fminf(fminf(fminf(c0, c1), fminf(c2, c3)),
                   fminf(fminf(c4, c5), fminf(c6, c7)));
  if (mn < t4) {
    ins5(c0, t0, t1, t2, t3, t4); ins5(c1, t0, t1, t2, t3, t4);
    ins5(c2, t0, t1, t2, t3, t4); ins5(c3, t0, t1, t2, t3, t4);
    ins5(c4, t0, t1, t2, t3, t4); ins5(c5, t0, t1, t2, t3, t4);
    ins5(c6, t0, t1, t2, t3, t4); ins5(c7, t0, t1, t2, t3, t4);
  }
}

// ---------------- kernel 0a: f32 -> f16 copy of x ----------------
__global__ __launch_bounds__(256) void k_convert(const float* __restrict__ x,
                                                 _Float16* __restrict__ xh) {
  int i = blockIdx.x * 256 + threadIdx.x;
  if (i < N_V * DIM) xh[i] = (_Float16)x[i];
}

// ---------------- kernel 0b: row squared norms (exact f32) ----------------
__global__ __launch_bounds__(256) void k_sq(const float* __restrict__ x,
                                            float* __restrict__ sq) {
  int i = blockIdx.x * 256 + threadIdx.x;
  if (i >= N_V) return;
  const float* p = x + i * DIM;
  float s = 0.f;
#pragma unroll
  for (int t = 0; t < DIM; ++t) s = fmaf(p[t], p[t], s);
  sq[i] = s;
}

// ---------------- kernel 1: vertex filtration via WMMA Gram + top-5 ----------------
// Block = 128 threads (4 waves). Block b owns rows [16b, 16b+16).
// Wave w scans column tiles c = w, w+4, ..., w+764 (192 tiles each).
__global__ __launch_bounds__(128) void k_vert(const _Float16* __restrict__ xh,
                                              const float* __restrict__ sq,
                                              float* __restrict__ vert,
                                              float* __restrict__ out) {
  __shared__ __align__(16) float tile[4][16 * LDS_STRIDE]; // per-wave d^2 tile
  __shared__ float cand[4][16][5];                         // per-wave per-row top-5

  const int lane    = threadIdx.x & 31;
  const int wv      = threadIdx.x >> 5;
  const int m       = lane & 15;       // row (scan) / column (B,store) index in tile
  const int hi      = lane >> 4;       // wave half
  const int hi8     = hi << 3;
  const int rowbase = blockIdx.x << 4;

  // Iteration at which this wave's column tile is the diagonal tile (scalar).
  const int diag_it = __builtin_amdgcn_readfirstlane(
      ((int)(blockIdx.x & 3) == wv) ? (int)(blockIdx.x >> 2) : -1);

  // --- A tile: rows rowbase..+15, K=0..31, contiguous K order (matches B):
  // lanes 0-15: halves 0-15 -> K 0-15
  // lanes 16-31: halves 0-15 -> K 16-31
  const _Float16* pA = xh + (rowbase + m) * DIM;
  v16h A = *(const v16h*)(pA + (hi << 4));

  // squared norms of the 8 rows this lane's accumulator covers (M = v + hi8)
  float sqr[8];
#pragma unroll
  for (int v = 0; v < 8; ++v) sqr[v] = sq[rowbase + hi8 + v];

  // running 5 smallest d^2 (ascending) for (row m, column-half hi)
  float t0 = FLT_BIG, t1 = FLT_BIG, t2 = FLT_BIG, t3 = FLT_BIG, t4 = FLT_BIG;

  float* myTile = tile[wv];

  // One column tile: WMMA -> d^2 -> LDS transpose -> top-5 scan.
  auto processTile = [&](v16h B, float sqj, bool isDiag) {
    v8f cz = {};
    v8f acc = __builtin_amdgcn_wmma_f32_16x16x32_f16(
        /*neg_a=*/false, A, /*neg_b=*/false, B,
        /*c_mod=*/(short)0, cz, /*reuse_a=*/false, /*reuse_b=*/false);

    // d^2 = |xi|^2 + |xj|^2 - 2 dot ; lane holds column n=m for rows v+hi8
#pragma unroll
    for (int v = 0; v < 8; ++v) {
      float d2 = sqr[v] + sqj - 2.0f * acc[v];
      myTile[(v + hi8) * LDS_STRIDE + m] = d2;
    }
    __builtin_amdgcn_wave_barrier();   // compiler fence; DS in-order per wave

    // scan: lane owns row m, columns hi8..hi8+7 (two ds_load_b128)
    v4f clo = *(const v4f*)(myTile + m * LDS_STRIDE + hi8);
    v4f chi = *(const v4f*)(myTile + m * LDS_STRIDE + hi8 + 4);

    if (!isDiag) {                     // scalar branch: fast path
      scan8(clo[0], clo[1], clo[2], clo[3], chi[0], chi[1], chi[2], chi[3],
            t0, t1, t2, t3, t4);
    } else {                           // diagonal tile: mask out self by index
      float d0 = (hi8 + 0 == m) ? FLT_BIG : clo[0];
      float d1 = (hi8 + 1 == m) ? FLT_BIG : clo[1];
      float d2_ = (hi8 + 2 == m) ? FLT_BIG : clo[2];
      float d3 = (hi8 + 3 == m) ? FLT_BIG : clo[3];
      float d4 = (hi8 + 4 == m) ? FLT_BIG : chi[0];
      float d5 = (hi8 + 5 == m) ? FLT_BIG : chi[1];
      float d6 = (hi8 + 6 == m) ? FLT_BIG : chi[2];
      float d7 = (hi8 + 7 == m) ? FLT_BIG : chi[3];
      scan8(d0, d1, d2_, d3, d4, d5, d6, d7, t0, t1, t2, t3, t4);
    }
    __builtin_amdgcn_wave_barrier();   // keep next stores after these loads
  };

  // --- B tiles: contiguous K order (matches A): lane holds column n=m,
  // halves 0-15 -> K = hi*16 .. hi*16+15 (contiguous in row-major x).
  // Register double-buffer + constant 4096B pointer stride; last iter peeled.
  // unroll 2: ping-pong the two B buffers so the Bcur<-Bnext copies rename away.
  int sqi = (wv << 4) + m;                         // B row / sq index
  const _Float16* pB = xh + sqi * DIM + (hi << 4);
  v16h  Bcur   = *(const v16h*)pB;
  float sqjcur = sq[sqi];

#pragma unroll 2
  for (int it = 0; it < NTILE - 1; ++it) {
    pB  += 64 * DIM;                               // +4 tiles = +4096 bytes
    sqi += 64;
    v16h  Bnext   = *(const v16h*)pB;
    float sqjnext = sq[sqi];

    processTile(Bcur, sqjcur, it == diag_it);

    Bcur = Bnext;
    sqjcur = sqjnext;
  }
  processTile(Bcur, sqjcur, (NTILE - 1) == diag_it);

  // --- merge column halves (lane L with lane L^16) via ds_bpermute
  {
    const int paddr = (lane ^ 16) << 2;
    float b0 = __int_as_float(__builtin_amdgcn_ds_bpermute(paddr, __float_as_int(t0)));
    float b1 = __int_as_float(__builtin_amdgcn_ds_bpermute(paddr, __float_as_int(t1)));
    float b2 = __int_as_float(__builtin_amdgcn_ds_bpermute(paddr, __float_as_int(t2)));
    float b3 = __int_as_float(__builtin_amdgcn_ds_bpermute(paddr, __float_as_int(t3)));
    float b4 = __int_as_float(__builtin_amdgcn_ds_bpermute(paddr, __float_as_int(t4)));
    float r0, r1, r2, r3, r4;
    merge5(t0, t1, t2, t3, t4, b0, b1, b2, b3, b4, r0, r1, r2, r3, r4);
    if (lane < 16) {
      cand[wv][m][0] = r0; cand[wv][m][1] = r1; cand[wv][m][2] = r2;
      cand[wv][m][3] = r3; cand[wv][m][4] = r4;
    }
  }
  __syncthreads();

  // --- wave 0 lanes 0..15: 4-way merge across waves, finalize vertex value
  if (threadIdx.x < 16) {
    const int row = threadIdx.x;
    float p0, p1, p2, p3, p4, q0, q1, q2, q3, q4, f0, f1, f2, f3, f4;
    merge5(cand[0][row][0], cand[0][row][1], cand[0][row][2], cand[0][row][3], cand[0][row][4],
           cand[1][row][0], cand[1][row][1], cand[1][row][2], cand[1][row][3], cand[1][row][4],
           p0, p1, p2, p3, p4);
    merge5(cand[2][row][0], cand[2][row][1], cand[2][row][2], cand[2][row][3], cand[2][row][4],
           cand[3][row][0], cand[3][row][1], cand[3][row][2], cand[3][row][3], cand[3][row][4],
           q0, q1, q2, q3, q4);
    merge5(p0, p1, p2, p3, p4, q0, q1, q2, q3, q4, f0, f1, f2, f3, f4);

    float s = expf(-sqrtf(fmaxf(f0, EPS_F)))
            + expf(-sqrtf(fmaxf(f1, EPS_F)))
            + expf(-sqrtf(fmaxf(f2, EPS_F)))
            + expf(-sqrtf(fmaxf(f3, EPS_F)))
            + expf(-sqrtf(fmaxf(f4, EPS_F)));
    const float vval = 1.0f - s * (1.0f / 6.0f);   // reference divides by NN_K=6
    const int ig = rowbase + row;
    vert[ig] = vval;
    out[2 * ig]     = vval;
    out[2 * ig + 1] = 0.0f;
  }
}

// ---------------- kernel 2: edge filtration ----------------
__global__ __launch_bounds__(256) void k_edge(const float* __restrict__ x,
                                              const int* __restrict__ ei,
                                              const float* __restrict__ vert,
                                              float* __restrict__ out) {
  int e = blockIdx.x * 256 + threadIdx.x;
  if (e >= N_E) return;
  const int u = ei[e];
  const int w = ei[N_E + e];
  const float* pu = x + u * DIM;
  const float* pw = x + w * DIM;
  float s = 0.f;
#pragma unroll
  for (int t = 0; t < DIM; ++t) {
    float d = pu[t] - pw[t];
    s = fmaf(d, d, s);
  }
  const float enorm = sqrtf(fmaxf(s, EPS_F));
  const float ey = 1.0f - expf(-enorm);
  const float ev = fmaxf(vert[u], vert[w]);   // symmetric -> pair sort irrelevant
  out[2 * (N_V + e)]     = ev;
  out[2 * (N_V + e) + 1] = ey;
}

// ---------------- launcher ----------------
extern "C" void kernel_launch(void* const* d_in, const int* in_sizes, int n_in,
                              void* d_out, int out_size, void* d_ws, size_t ws_size,
                              hipStream_t stream) {
  const float* x  = (const float*)d_in[0];
  const int*   ei = (const int*)d_in[1];
  float* out = (float*)d_out;

  char* ws = (char*)d_ws;
  _Float16* xh  = (_Float16*)ws;                                        // 768 KB
  float*    sq  = (float*)(ws + (size_t)N_V * DIM * 2);                 // 48 KB
  float*    vrt = (float*)(ws + (size_t)N_V * DIM * 2 + (size_t)N_V * 4);

  k_convert<<<(N_V * DIM + 255) / 256, 256, 0, stream>>>(x, xh);
  k_sq<<<(N_V + 255) / 256, 256, 0, stream>>>(x, sq);
  k_vert<<<N_V / 16, 128, 0, stream>>>(xh, sq, vrt, out);
  k_edge<<<N_E / 256, 256, 0, stream>>>(x, ei, vrt, out);
}
